// SelectiveScan_71640054497679
// MI455X (gfx1250) — matrix operand order repaired and checked by reading, not compiled, not run on hardware
//
#include <hip/hip_runtime.h>
#include <math.h>

// Model constants from the reference
#define DMODEL 1024
#define DINNER 2048
#define DSTATE 16
#define DCONV  4
#define SEQL   2048
#define NCHUNK 16
#define CHUNK  (SEQL / NCHUNK)   // 128

typedef __attribute__((ext_vector_type(16))) _Float16 v16h;
typedef __attribute__((ext_vector_type(8)))  _Float16 v8h;
typedef __attribute__((ext_vector_type(8)))  float    v8f;

// ---------------------------------------------------------------------------
// fp32 -> fp16 conversion (for feeding WMMA)
// ---------------------------------------------------------------------------
__global__ void __launch_bounds__(256) cvt_f32_to_f16_kernel(
    const float* __restrict__ in, _Float16* __restrict__ out, int n)
{
    int i = blockIdx.x * blockDim.x + threadIdx.x;
    if (i < n) out[i] = (_Float16)in[i];
}

// ---------------------------------------------------------------------------
// WMMA fragment loaders (layouts per CDNA5 ISA, wave32)
// A 16x32 f16: lane r holds row r; elems 0..7 = K[k0+8h..], 8..15 = K[k0+16+8h..]
// B 32x16 f16 (from row-major W[n][k]): lane gives N; elems 0..7 = K[k0+8h..], 8..15 = K[k0+16+8h..]
// ---------------------------------------------------------------------------
__device__ __forceinline__ v16h load_a_frag(const _Float16* rowPtr, int k0, int hl)
{
    const v8h a0 = *(const v8h*)(rowPtr + k0 + hl * 8);
    const v8h a1 = *(const v8h*)(rowPtr + k0 + 16 + hl * 8);
    v16h a;
    #pragma unroll
    for (int i = 0; i < 8; ++i) { a[i] = a0[i]; a[8 + i] = a1[i]; }
    return a;
}

__device__ __forceinline__ v16h load_b_frag(const _Float16* rowPtr, int k0, int hl)
{
    const _Float16* p = rowPtr + k0 + hl * 8;
    const v8h b0 = *(const v8h*)(p);
    const v8h b1 = *(const v8h*)(p + 16);
    v16h b;
    #pragma unroll
    for (int i = 0; i < 8; ++i) { b[i] = b0[i]; b[8 + i] = b1[i]; }
    return b;
}

#define WMMA_F16(A_, B_, C_) __builtin_amdgcn_wmma_f32_16x16x32_f16( \
    false, (A_), false, (B_), (short)0, (C_), false, false)

// ---------------------------------------------------------------------------
// WMMA GEMM: C[M,N] (fp32) = A[M,K] (f16, ld=lda) * W[N,K]^T (f16, ld=ldw)
// Each wave computes a 32x64 block (2 M-tiles x 4 N-tiles, 8 accumulators).
// K double-buffered in registers (unroll-by-64 ping-pong) so loads for the
// next K-step overlap the 8 WMMAs of the current step (partial loadcnt waits).
// Requirements: M%32==0, K%64==0.  N arbitrary (clamped loads, masked stores).
// ---------------------------------------------------------------------------
__global__ void __launch_bounds__(256) wmma_gemm_f16_kernel(
    const _Float16* __restrict__ A, const _Float16* __restrict__ W,
    float* __restrict__ C, int M, int N, int K, int lda, int ldw, int ldc)
{
    const int wavesPerBlock = blockDim.x >> 5;
    const int gw = blockIdx.x * wavesPerBlock + (int)(threadIdx.x >> 5);
    const int numN64 = (N + 63) >> 6;
    const int totalWaves = (M >> 5) * numN64;
    if (gw >= totalWaves) return;   // wave-uniform exit: EXEC stays all-ones

    const int mTile = gw / numN64;
    const int n64   = gw - mTile * numN64;
    const int mBase = mTile << 5;
    const int nBase = n64 << 6;

    const int lane = threadIdx.x & 31;
    const int hl   = lane >> 4;
    const int r    = lane & 15;

    const _Float16* aRow0 = A + (size_t)(mBase + r) * lda;
    const _Float16* aRow1 = A + (size_t)(mBase + 16 + r) * lda;
    const _Float16* wRow[4];
    #pragma unroll
    for (int t = 0; t < 4; ++t) {
        int n  = nBase + t * 16 + r;
        int nc = n < N ? n : N - 1;            // clamp OOB rows (stores masked)
        wRow[t] = W + (size_t)nc * ldw;
    }

    v8f acc[8];
    #pragma unroll
    for (int i = 0; i < 8; ++i) acc[i] = (v8f){0.f,0.f,0.f,0.f,0.f,0.f,0.f,0.f};

    // prologue: fill buffer A at k=0
    v16h a0A = load_a_frag(aRow0, 0, hl);
    v16h a1A = load_a_frag(aRow1, 0, hl);
    v16h b0A = load_b_frag(wRow[0], 0, hl);
    v16h b1A = load_b_frag(wRow[1], 0, hl);
    v16h b2A = load_b_frag(wRow[2], 0, hl);
    v16h b3A = load_b_frag(wRow[3], 0, hl);

    for (int k0 = 0; k0 < K; k0 += 64) {
        // ---- stage even: prefetch buffer B at k0+32, compute on buffer A ----
        const int k1 = k0 + 32;                 // always < K (K%64==0)
        v16h a0B = load_a_frag(aRow0, k1, hl);
        v16h a1B = load_a_frag(aRow1, k1, hl);
        v16h b0B = load_b_frag(wRow[0], k1, hl);
        v16h b1B = load_b_frag(wRow[1], k1, hl);
        v16h b2B = load_b_frag(wRow[2], k1, hl);
        v16h b3B = load_b_frag(wRow[3], k1, hl);

        acc[0] = WMMA_F16(a0A, b0A, acc[0]);
        acc[1] = WMMA_F16(a1A, b0A, acc[1]);
        acc[2] = WMMA_F16(a0A, b1A, acc[2]);
        acc[3] = WMMA_F16(a1A, b1A, acc[3]);
        acc[4] = WMMA_F16(a0A, b2A, acc[4]);
        acc[5] = WMMA_F16(a1A, b2A, acc[5]);
        acc[6] = WMMA_F16(a0A, b3A, acc[6]);
        acc[7] = WMMA_F16(a1A, b3A, acc[7]);

        // ---- stage odd: prefetch buffer A at k0+64 (dummy reload on last) ---
        const int k2 = (k0 + 64 < K) ? (k0 + 64) : k0;
        a0A = load_a_frag(aRow0, k2, hl);
        a1A = load_a_frag(aRow1, k2, hl);
        b0A = load_b_frag(wRow[0], k2, hl);
        b1A = load_b_frag(wRow[1], k2, hl);
        b2A = load_b_frag(wRow[2], k2, hl);
        b3A = load_b_frag(wRow[3], k2, hl);

        acc[0] = WMMA_F16(a0B, b0B, acc[0]);
        acc[1] = WMMA_F16(a1B, b0B, acc[1]);
        acc[2] = WMMA_F16(a0B, b1B, acc[2]);
        acc[3] = WMMA_F16(a1B, b1B, acc[3]);
        acc[4] = WMMA_F16(a0B, b2B, acc[4]);
        acc[5] = WMMA_F16(a1B, b2B, acc[5]);
        acc[6] = WMMA_F16(a0B, b3B, acc[6]);
        acc[7] = WMMA_F16(a1B, b3B, acc[7]);
    }

    // C/D layout: VGPR i -> M = tileBase + 8*hl + i, N = nBase + 16t + r
    #pragma unroll
    for (int t = 0; t < 4; ++t) {
        int n = nBase + t * 16 + r;
        if (n < N) {
            #pragma unroll
            for (int mi = 0; mi < 2; ++mi) {
                int mRow = mBase + mi * 16 + hl * 8;
                float* cp = C + (size_t)mRow * ldc + n;
                #pragma unroll
                for (int i = 0; i < 8; ++i) cp[(size_t)i * ldc] = acc[t * 2 + mi][i];
            }
        }
    }
}

// ---------------------------------------------------------------------------
// Depthwise causal conv (width 4) + bias + SiLU.  xi lives in xz[:, 0:2048]
// with row stride 4096.  Writes xc (fp32) and xc_h (f16 copy for WMMA).
// ---------------------------------------------------------------------------
__global__ void __launch_bounds__(256) conv_silu_kernel(
    const float* __restrict__ xz, const float* __restrict__ conv_w,
    const float* __restrict__ conv_b, float* __restrict__ xc,
    _Float16* __restrict__ xc_h)
{
    int idx = blockIdx.x * blockDim.x + threadIdx.x;
    if (idx >= SEQL * DINNER) return;
    int l = idx >> 11;
    int d = idx & (DINNER - 1);
    float s = conv_b[d];
    #pragma unroll
    for (int j = 0; j < DCONV; ++j) {
        int t = l - (DCONV - 1) + j;
        if (t >= 0) s += conv_w[d * DCONV + j] * xz[(size_t)t * (2 * DINNER) + d];
    }
    float sv = s / (1.0f + expf(-s));   // SiLU
    xc[idx]   = sv;
    xc_h[idx] = (_Float16)sv;
}

// ---------------------------------------------------------------------------
// Chunked selective scan, pass 1: per-(chunk, channel) summaries.
// Computes prodA[n] = prod_t exp(dt*A[n]) and the local final state assuming
// a zero entry state.  16 chunks of 128 steps -> 16x the parallelism of a
// monolithic scan.
// ---------------------------------------------------------------------------
__device__ __forceinline__ float softplus_clip(float dp)
{
    float delta = (dp > 20.0f) ? dp : log1pf(expf(dp));
    return fminf(fmaxf(delta, -10.0f), 10.0f);
}

__global__ void __launch_bounds__(256) scan_pass1_kernel(
    const float* __restrict__ delta_pre,  // (L, DINNER)
    const float* __restrict__ x_dbl,      // (L, 32); B_ssm = [:,16:32]
    const float* __restrict__ xc,         // (L, DINNER)
    const float* __restrict__ A_log,      // (DINNER, 16)
    const float* __restrict__ b_dt,       // (DINNER)
    float* __restrict__ prodA,            // (NCHUNK, DINNER, 16)
    float* __restrict__ hloc)             // (NCHUNK, DINNER, 16)
{
    int idx = blockIdx.x * blockDim.x + threadIdx.x;
    if (idx >= NCHUNK * DINNER) return;
    int c = idx >> 11;            // / DINNER
    int d = idx & (DINNER - 1);

    float A[DSTATE], pA[DSTATE], h[DSTATE];
    #pragma unroll
    for (int n = 0; n < DSTATE; ++n) {
        A[n]  = -expf(A_log[(size_t)d * DSTATE + n]);
        pA[n] = 1.0f;
        h[n]  = 0.0f;
    }
    const float bdt = b_dt[d];

    const int t0 = c * CHUNK;
    for (int t = t0; t < t0 + CHUNK; ++t) {
        float delta = softplus_clip(delta_pre[(size_t)t * DINNER + d] + bdt);
        float xv = xc[(size_t)t * DINNER + d];
        const float* Bs = x_dbl + (size_t)t * 32 + DSTATE;
        #pragma unroll
        for (int n = 0; n < DSTATE; ++n) {
            float a = expf(delta * A[n]);
            pA[n] *= a;
            h[n] = a * h[n] + (delta * Bs[n]) * xv;
        }
    }
    float* pAo = prodA + (size_t)idx * DSTATE;
    float* hlo = hloc  + (size_t)idx * DSTATE;
    #pragma unroll
    for (int n = 0; n < DSTATE; ++n) { pAo[n] = pA[n]; hlo[n] = h[n]; }
}

// ---------------------------------------------------------------------------
// Chunked scan, combine: propagate entry states across the 16 chunks.
// One lane per (channel, state): 32768 lanes, 16-step loop.
// ---------------------------------------------------------------------------
__global__ void __launch_bounds__(256) scan_combine_kernel(
    const float* __restrict__ prodA, const float* __restrict__ hloc,
    float* __restrict__ hentry)           // (NCHUNK, DINNER, 16)
{
    int idx = blockIdx.x * blockDim.x + threadIdx.x;
    if (idx >= DINNER * DSTATE) return;   // idx = d*16 + n
    float h = 0.0f;
    #pragma unroll
    for (int c = 0; c < NCHUNK; ++c) {
        size_t o = (size_t)c * DINNER * DSTATE + idx;
        hentry[o] = h;
        h = prodA[o] * h + hloc[o];
    }
}

// ---------------------------------------------------------------------------
// Chunked scan, pass 2: re-run each chunk from its correct entry state,
// fully parallel across chunks; fuses D-skip and SiLU(z) gate, writes y (f16).
// ---------------------------------------------------------------------------
__global__ void __launch_bounds__(256) scan_pass2_kernel(
    const float* __restrict__ delta_pre,
    const float* __restrict__ x_dbl,
    const float* __restrict__ xc,
    const float* __restrict__ xz,         // z = [:, 2048:4096], ld 4096
    const float* __restrict__ A_log,
    const float* __restrict__ D_param,
    const float* __restrict__ b_dt,
    const float* __restrict__ hentry,
    _Float16* __restrict__ y_h)           // (L, DINNER)
{
    int idx = blockIdx.x * blockDim.x + threadIdx.x;
    if (idx >= NCHUNK * DINNER) return;
    int c = idx >> 11;
    int d = idx & (DINNER - 1);

    float A[DSTATE], h[DSTATE];
    const float* he = hentry + (size_t)idx * DSTATE;
    #pragma unroll
    for (int n = 0; n < DSTATE; ++n) {
        A[n] = -expf(A_log[(size_t)d * DSTATE + n]);
        h[n] = he[n];
    }
    const float bdt = b_dt[d];
    const float Dp  = D_param[d];

    const int t0 = c * CHUNK;
    for (int t = t0; t < t0 + CHUNK; ++t) {
        float delta = softplus_clip(delta_pre[(size_t)t * DINNER + d] + bdt);
        float xv = xc[(size_t)t * DINNER + d];
        const float* Bs = x_dbl + (size_t)t * 32 + DSTATE;
        float acc = 0.0f;
        #pragma unroll
        for (int n = 0; n < DSTATE; ++n) {
            float a = expf(delta * A[n]);
            h[n] = a * h[n] + (delta * Bs[n]) * xv;
            acc += h[n];
        }
        acc += xv * Dp;
        float zv = xz[(size_t)t * (2 * DINNER) + DINNER + d];
        float yv = acc * (zv / (1.0f + expf(-zv)));
        y_h[(size_t)t * DINNER + d] = (_Float16)yv;
    }
}

// ---------------------------------------------------------------------------
// Host-side launch
// ---------------------------------------------------------------------------
static inline int gemm_blocks(int M, int N) {
    int numN64 = (N + 63) >> 6;
    int totalWaves = (M >> 5) * numN64;   // 32x64 per wave
    return (totalWaves + 7) / 8;          // 8 waves / 256-thread block
}

extern "C" void kernel_launch(void* const* d_in, const int* in_sizes, int n_in,
                              void* d_out, int out_size, void* d_ws, size_t ws_size,
                              hipStream_t stream) {
    const float* x      = (const float*)d_in[0];
    const float* W_in   = (const float*)d_in[1];
    const float* conv_w = (const float*)d_in[2];
    const float* conv_b = (const float*)d_in[3];
    const float* W_x    = (const float*)d_in[4];
    const float* W_dt   = (const float*)d_in[5];
    const float* b_dt   = (const float*)d_in[6];
    const float* A_log  = (const float*)d_in[7];
    const float* D_par  = (const float*)d_in[8];
    const float* W_out  = (const float*)d_in[9];
    float* out = (float*)d_out;

    char* ws = (char*)d_ws;
    size_t off = 0;
    auto take = [&](size_t bytes) -> char* {
        char* p = ws + off;
        off = (off + bytes + 255) & ~(size_t)255;
        return p;
    };
    _Float16* x_h     = (_Float16*)take((size_t)SEQL * DMODEL * 2);
    _Float16* W_in_h  = (_Float16*)take((size_t)2 * DINNER * DMODEL * 2);
    _Float16* W_x_h   = (_Float16*)take((size_t)32 * DINNER * 2);
    _Float16* W_dt_h  = (_Float16*)take((size_t)DINNER * DINNER * 2);
    _Float16* W_out_h = (_Float16*)take((size_t)DMODEL * DINNER * 2);
    float*    xz      = (float*)   take((size_t)SEQL * 2 * DINNER * 4);
    float*    xc      = (float*)   take((size_t)SEQL * DINNER * 4);
    _Float16* xc_h    = (_Float16*)take((size_t)SEQL * DINNER * 2);
    float*    x_dbl   = (float*)   take((size_t)SEQL * 32 * 4);
    float*    dpre    = (float*)   take((size_t)SEQL * DINNER * 4);
    _Float16* y_h     = (_Float16*)take((size_t)SEQL * DINNER * 2);
    float*    prodA   = (float*)   take((size_t)NCHUNK * DINNER * DSTATE * 4);
    float*    hloc    = (float*)   take((size_t)NCHUNK * DINNER * DSTATE * 4);
    float*    hentry  = (float*)   take((size_t)NCHUNK * DINNER * DSTATE * 4);
    (void)ws_size; (void)n_in; (void)in_sizes; (void)out_size;

    // ---- 1. fp16 copies of GEMM operands ----
    auto cvt = [&](const float* src, _Float16* dst, int n) {
        cvt_f32_to_f16_kernel<<<(n + 255) / 256, 256, 0, stream>>>(src, dst, n);
    };
    cvt(x,     x_h,     SEQL * DMODEL);
    cvt(W_in,  W_in_h,  2 * DINNER * DMODEL);
    cvt(W_x,   W_x_h,   32 * DINNER);
    cvt(W_dt,  W_dt_h,  DINNER * DINNER);
    cvt(W_out, W_out_h, DMODEL * DINNER);

    // ---- 2. xz = x @ W_in^T   (M=2048, N=4096, K=1024) ----
    wmma_gemm_f16_kernel<<<gemm_blocks(SEQL, 2 * DINNER), 256, 0, stream>>>(
        x_h, W_in_h, xz, SEQL, 2 * DINNER, DMODEL, DMODEL, DMODEL, 2 * DINNER);

    // ---- 3. depthwise conv + SiLU -> xc (fp32 + f16) ----
    conv_silu_kernel<<<(SEQL * DINNER + 255) / 256, 256, 0, stream>>>(
        xz, conv_w, conv_b, xc, xc_h);

    // ---- 4. x_dbl = xc @ W_x^T   (N=32; B_ssm = cols 16..31) ----
    wmma_gemm_f16_kernel<<<gemm_blocks(SEQL, 32), 256, 0, stream>>>(
        xc_h, W_x_h, x_dbl, SEQL, 32, DINNER, DINNER, DINNER, 32);

    // ---- 5. delta_pre = xc @ W_dt^T   (M=N=K=2048) ----
    wmma_gemm_f16_kernel<<<gemm_blocks(SEQL, DINNER), 256, 0, stream>>>(
        xc_h, W_dt_h, dpre, SEQL, DINNER, DINNER, DINNER, DINNER, DINNER);

    // ---- 6. chunked selective scan -> y_h (f16) ----
    scan_pass1_kernel<<<(NCHUNK * DINNER + 255) / 256, 256, 0, stream>>>(
        dpre, x_dbl, xc, A_log, b_dt, prodA, hloc);
    scan_combine_kernel<<<(DINNER * DSTATE + 255) / 256, 256, 0, stream>>>(
        prodA, hloc, hentry);
    scan_pass2_kernel<<<(NCHUNK * DINNER + 255) / 256, 256, 0, stream>>>(
        dpre, x_dbl, xc, xz, A_log, D_par, b_dt, hentry, y_h);

    // ---- 7. out = y @ W_out^T   (M=2048, N=1024, K=2048) ----
    wmma_gemm_f16_kernel<<<gemm_blocks(SEQL, DMODEL), 256, 0, stream>>>(
        y_h, W_out_h, out, SEQL, DMODEL, DINNER, DINNER, DINNER, DMODEL);
}
